// MinGRU_82162724372985
// MI455X (gfx1250) — matrix operand order repaired and checked by reading, not compiled, not run on hardware
//
#include <hip/hip_runtime.h>
#include <stdint.h>

// ---------------- problem constants ----------------
constexpr int  CB = 128;            // batch
constexpr int  CT = 1024;           // time
constexpr int  CI = 256;            // input dim
constexpr int  CH = 512;            // hidden dim
constexpr long long CBT = (long long)CB * CT;   // 131072 rows
constexpr int  HSTR = 520;          // padded LDS row stride (halves) -> bank-conflict-free b128 reads

// ---------------- WMMA plumbing ----------------
typedef float  v8f   __attribute__((ext_vector_type(8)));
typedef __bf16 v16bf __attribute__((ext_vector_type(16)));

union Frag { uint4 u[2]; v16bf v; };

static __device__ __forceinline__ v8f wmma_bf(const Frag& a, const Frag& b, v8f c) {
  // D = A(16x32 bf16) * B(32x16 bf16) + C(16x16 f32)
  return __builtin_amdgcn_wmma_f32_16x16x32_bf16(false, a.v, false, b.v, (short)0, c, false, false);
}

static __device__ __forceinline__ unsigned short f2bf(float f) {
  union { float f; unsigned u; } x; x.f = f;
  unsigned r = x.u + 0x7FFFu + ((x.u >> 16) & 1u);   // round-to-nearest-even
  return (unsigned short)(r >> 16);
}

static __device__ __forceinline__ float fsigmoid(float x) {
  return 1.0f / (1.0f + __expf(-x));
}
static __device__ __forceinline__ float ftanh(float x) {
  // tanh(x) = 1 - 2/(exp(2x)+1); saturates correctly for |x| large
  return 1.0f - 2.0f / (__expf(2.0f * x) + 1.0f);
}

// ---------------- kernel 1: f32 -> bf16 copy of x ----------------
__global__ void k_cvt_x(const float* __restrict__ x, unsigned short* __restrict__ xbf, long long n) {
  long long i = ((long long)blockIdx.x * blockDim.x + threadIdx.x) * 8;
#pragma unroll
  for (int j = 0; j < 8; ++j)
    if (i + j < n) xbf[i + j] = f2bf(x[i + j]);
}

// ---------------- kernel 2: pack a (Kdim x H) f32 weight into bf16 B-fragment order ----
// Fragment (nt,kt) is 512 halves at ((nt*NK)+kt)*512; within: element l*16+j = W[kt*32+8*(l>>4)+(j&7)+16*(j>>3)][nt*16+(l&15)]
// (lane%16 = N index, halves = K index in WMMA table order -> a wave's fragment load is 1KB contiguous).
__global__ void k_pack(const float* __restrict__ src, int Kdim, unsigned short* __restrict__ dst) {
  int e = blockIdx.x * blockDim.x + threadIdx.x;
  int total = Kdim * CH;
  if (e >= total) return;
  int NK = Kdim / 32;
  int f = e >> 9, rem = e & 511;
  int l = rem >> 4, j = rem & 15;
  int kt = f % NK, nt = f / NK;
  int k = kt * 32 + 8 * (l >> 4) + (j & 7) + 16 * (j >> 3), n = nt * 16 + (l & 15);
  dst[e] = f2bf(src[(long long)k * CH + n]);
}

// ---------------- kernel 3: fused precompute GEMM  xz = x@Wzx + bz ; xh = x@Whx + bh ----
// wave tile: 32(M) x 32(N), both output matrices. grid covers (BT/32) x (H/32) waves exactly.
__global__ void __launch_bounds__(256) k_gemm_pre(
    const unsigned short* __restrict__ xbf,
    const unsigned short* __restrict__ wzx, const unsigned short* __restrict__ whx,
    const float* __restrict__ bz, const float* __restrict__ bh,
    float* __restrict__ xz, float* __restrict__ xh) {
  const int lane = threadIdx.x & 31;
  const int wv   = threadIdx.x >> 5;
  const int gw   = blockIdx.x * 8 + wv;
  const int mt   = gw >> 4;        // BT/32 = 4096 row-tiles
  const int nq   = gw & 15;        // H/32  = 16 col-tiles
  const int rowBase = mt * 32;
  const int colBase = nq * 32;
  const int lm = lane & 15;
  const int hiOff = (lane >= 16) ? 8 : 0;

  v8f acc[2][2][2] = {};           // [mi][ni][z/h]
  constexpr int NKI = CI / 32;     // 8

  for (int kt = 0; kt < NKI; ++kt) {
    Frag A[2];
#pragma unroll
    for (int mi = 0; mi < 2; ++mi) {
      const unsigned short* rp = xbf + (long long)(rowBase + mi * 16 + lm) * CI + kt * 32 + hiOff;
      A[mi].u[0] = *(const uint4*)rp;
      A[mi].u[1] = *(const uint4*)(rp + 16);
    }
    Frag Bz[2], Bh[2];
#pragma unroll
    for (int ni = 0; ni < 2; ++ni) {
      int f = ((colBase >> 4) + ni) * NKI + kt;
      const uint4* pz = (const uint4*)(wzx + (long long)f * 512) + lane * 2;
      const uint4* ph = (const uint4*)(whx + (long long)f * 512) + lane * 2;
      Bz[ni].u[0] = pz[0]; Bz[ni].u[1] = pz[1];
      Bh[ni].u[0] = ph[0]; Bh[ni].u[1] = ph[1];
    }
#pragma unroll
    for (int mi = 0; mi < 2; ++mi)
#pragma unroll
      for (int ni = 0; ni < 2; ++ni) {
        acc[mi][ni][0] = wmma_bf(A[mi], Bz[ni], acc[mi][ni][0]);
        acc[mi][ni][1] = wmma_bf(A[mi], Bh[ni], acc[mi][ni][1]);
      }
  }
  // C layout: VGPR r, lanes 0-15 -> (M=r, N=lane); lanes 16-31 -> (M=r+8, N=lane-16)
#pragma unroll
  for (int mi = 0; mi < 2; ++mi)
#pragma unroll
    for (int ni = 0; ni < 2; ++ni) {
      int col = colBase + ni * 16 + lm;
      float bzv = bz[col], bhv = bh[col];
#pragma unroll
      for (int r = 0; r < 8; ++r) {
        int row = rowBase + mi * 16 + r + hiOff;
        long long off = (long long)row * CH + col;
        xz[off] = acc[mi][ni][0][r] + bzv;
        xh[off] = acc[mi][ni][1][r] + bhv;   // xh lives in d_out; overwritten by h_t in scan
      }
    }
}

// ---------------- kernel 4: sequential scan ----------------
// 4 workgroups, each owns 32 batch rows. 16 waves/WG; wave w owns columns [w*32,w*32+32).
// h state: bf16 ping-pong in LDS (feeds next step's A fragments), f32 carried in registers.
__global__ void __launch_bounds__(512) k_scan(
    const unsigned short* __restrict__ wzh, const unsigned short* __restrict__ whh,
    const float* __restrict__ xz, float* __restrict__ out) {
  __shared__ __align__(16) unsigned short hsh[2][32 * HSTR];

  const int lane = threadIdx.x & 31;
  const int wv   = threadIdx.x >> 5;
  const int bb   = blockIdx.x;          // 0..3
  const int colBase = wv * 32;
  const int lm = lane & 15;
  const int hiOff = (lane >= 16) ? 8 : 0;
  constexpr int NKH = CH / 32;          // 16

  for (int i = threadIdx.x; i < 32 * HSTR; i += 512) hsh[0][i] = 0;
  v8f hprev[2][2] = {};                 // f32 carried state, [mi][ni]
  __syncthreads();

  int p = 0;
  for (int t = 0; t < CT; ++t) {
    v8f acc[2][2][2] = {};
    for (int kt = 0; kt < NKH; ++kt) {
      Frag A[2];
#pragma unroll
      for (int mi = 0; mi < 2; ++mi) {
        const unsigned short* rp = &hsh[p][(mi * 16 + lm) * HSTR + kt * 32 + hiOff];
        A[mi].u[0] = *(const uint4*)rp;
        A[mi].u[1] = *(const uint4*)(rp + 16);
      }
      Frag Bz[2], Bh[2];
#pragma unroll
      for (int ni = 0; ni < 2; ++ni) {
        int f = ((colBase >> 4) + ni) * NKH + kt;
        const uint4* pz = (const uint4*)(wzh + (long long)f * 512) + lane * 2;
        const uint4* ph = (const uint4*)(whh + (long long)f * 512) + lane * 2;
        Bz[ni].u[0] = pz[0]; Bz[ni].u[1] = pz[1];
        Bh[ni].u[0] = ph[0]; Bh[ni].u[1] = ph[1];
      }
#pragma unroll
      for (int mi = 0; mi < 2; ++mi)
#pragma unroll
        for (int ni = 0; ni < 2; ++ni) {
          acc[mi][ni][0] = wmma_bf(A[mi], Bz[ni], acc[mi][ni][0]);
          acc[mi][ni][1] = wmma_bf(A[mi], Bh[ni], acc[mi][ni][1]);
        }
    }
    // elementwise gate + update; read xz (ws) and xh (in-place in out), write h_t
#pragma unroll
    for (int mi = 0; mi < 2; ++mi)
#pragma unroll
      for (int ni = 0; ni < 2; ++ni) {
        int col = colBase + ni * 16 + lm;
#pragma unroll
        for (int r = 0; r < 8; ++r) {
          int ml = mi * 16 + r + hiOff;              // local batch row 0..31
          int b  = bb * 32 + ml;
          long long off = ((long long)b * CT + t) * CH + col;
          float z  = fsigmoid(acc[mi][ni][0][r] + xz[off]);
          float ht = ftanh   (acc[mi][ni][1][r] + out[off]);
          float hp = hprev[mi][ni][r];
          float hn = hp + z * (ht - hp);
          out[off] = hn;
          hprev[mi][ni][r] = hn;
          hsh[p ^ 1][ml * HSTR + col] = f2bf(hn);
        }
      }
    __syncthreads();   // one barrier/step: ping-pong makes read-before-overwrite safe
    p ^= 1;
  }
  // h_last
#pragma unroll
  for (int mi = 0; mi < 2; ++mi)
#pragma unroll
    for (int ni = 0; ni < 2; ++ni) {
      int col = colBase + ni * 16 + lm;
#pragma unroll
      for (int r = 0; r < 8; ++r) {
        int b = bb * 32 + mi * 16 + r + hiOff;
        out[(long long)CBT * CH + (long long)b * CH + col] = hprev[mi][ni][r];
      }
    }
}

// ---------------- host launch ----------------
extern "C" void kernel_launch(void* const* d_in, const int* in_sizes, int n_in,
                              void* d_out, int out_size, void* d_ws, size_t ws_size,
                              hipStream_t stream) {
  const float* x  = (const float*)d_in[0];
  const float* Wz = (const float*)d_in[1];
  const float* bz = (const float*)d_in[2];
  const float* Wh = (const float*)d_in[3];
  const float* bh = (const float*)d_in[4];
  float* out = (float*)d_out;

  // workspace layout (~322 MB): x_bf16 | packed Wzx | Whx | Wzh | Whh | xz(f32)
  char* ws = (char*)d_ws;
  unsigned short* xbf = (unsigned short*)ws;
  size_t o = (size_t)CBT * CI * 2;                       // 64 MB
  unsigned short* wzxP = (unsigned short*)(ws + o); o += (size_t)CI * CH * 2;
  unsigned short* whxP = (unsigned short*)(ws + o); o += (size_t)CI * CH * 2;
  unsigned short* wzhP = (unsigned short*)(ws + o); o += (size_t)CH * CH * 2;
  unsigned short* whhP = (unsigned short*)(ws + o); o += (size_t)CH * CH * 2;
  float* xzbuf = (float*)(ws + o);                       // 256 MB

  long long nX = CBT * CI;
  k_cvt_x<<<(unsigned)((nX / 8 + 255) / 256), 256, 0, stream>>>(x, xbf, nX);

  k_pack<<<(CI * CH + 255) / 256, 256, 0, stream>>>(Wz,                      CI, wzxP);
  k_pack<<<(CI * CH + 255) / 256, 256, 0, stream>>>(Wh,                      CI, whxP);
  k_pack<<<(CH * CH + 255) / 256, 256, 0, stream>>>(Wz + (size_t)CI * CH,    CH, wzhP);
  k_pack<<<(CH * CH + 255) / 256, 256, 0, stream>>>(Wh + (size_t)CI * CH,    CH, whhP);

  // (BT/32)*(H/32) = 65536 waves, 8 waves/block
  k_gemm_pre<<<8192, 256, 0, stream>>>(xbf, wzxP, whxP, bz, bh, xzbuf, out);

  // 4 workgroups x 512 threads; sequential over T inside the kernel
  k_scan<<<4, 512, 0, stream>>>(wzhP, whhP, xzbuf, out);
}
